// ITASelfAttention_QAT_65549790871614
// MI455X (gfx1250) — compile-verified, matrix operand order checked by reading, not run on hardware
//
#include <hip/hip_runtime.h>

typedef unsigned short u16;
typedef unsigned int u32;
typedef unsigned long long u64;
typedef __attribute__((ext_vector_type(16))) __bf16 bf16x16;
typedef __attribute__((ext_vector_type(8)))  float  floatx8;

#define LOG2E_F 1.4426950408889634f

// ---------- sizes ----------
#define BB 8
#define NN 1024
#define EE 768
#define PP 768
#define HH 12
#define DD 64
#define BN (BB*NN)          // 8192

// ---------- workspace layout (bytes) ----------
#define OFF_AMAX 0
#define OFF_XQ   256                       // 8192*768 bf16
#define SZ_XQ    ((size_t)BN*EE*2)
#define OFF_WQT  (OFF_XQ + SZ_XQ)
#define SZ_W     ((size_t)EE*PP*2)
#define OFF_WKT  (OFF_WQT + SZ_W)
#define OFF_WVT  (OFF_WKT + SZ_W)
#define OFF_WOT  (OFF_WVT + SZ_W)
#define OFF_Q    (OFF_WOT + SZ_W)          // [B*H, N, D] bf16
#define SZ_QKV   ((size_t)BB*HH*NN*DD*2)
#define OFF_K    (OFF_Q + SZ_QKV)
#define OFF_VT   (OFF_K + SZ_QKV)          // [B*H, D, N] bf16
#define OFF_CTX  (OFF_VT + SZ_QKV)         // [8192, 768] bf16

// ---------- helpers ----------
__device__ __forceinline__ u16 f2bf(float x) {
  u32 u = __float_as_uint(x);
  u32 r = u + 0x7FFFu + ((u >> 16) & 1u);   // round to nearest even
  return (u16)(r >> 16);
}
__device__ __forceinline__ u32 pack_bf2(float lo, float hi) {
  return (u32)f2bf(lo) | ((u32)f2bf(hi) << 16);
}

union FragU { bf16x16 v; u32 u[8]; };

// A/B fragment for v_wmma_f32_16x16x32_bf16.
// Per ISA 16-bit 16x32 A layout: lane = half*16+m holds row m;
// VGPR v<4:  k = 8*half + 2v {,+1};  v>=4: k = 16 + 8*half + 2(v-4) {,+1}.
// B matrices are stored transposed ([N][K]) so lane n reads its column
// contiguously with the identical k mapping.
__device__ __forceinline__ bf16x16 load_frag_g(const u16* base, int ld,
                                               int row0, int k0, int lane) {
  int m = lane & 15, half = lane >> 4;
  const u16* p = base + (size_t)(row0 + m) * ld + k0 + 8 * half;
  FragU f;
#pragma unroll
  for (int v = 0; v < 4; ++v) f.u[v]     = *(const u32*)(p + 2 * v);
#pragma unroll
  for (int v = 0; v < 4; ++v) f.u[4 + v] = *(const u32*)(p + 16 + 2 * v);
  return f.v;
}

// fragment from LDS-resident bf16 tile (two contiguous 16B chunks per lane)
__device__ __forceinline__ bf16x16 load_frag_l(const u16* base, int ld,
                                               int k0, int lane) {
  int m = lane & 15, half = lane >> 4;
  const u16* p = base + m * ld + k0 + 8 * half;
  FragU f;
#pragma unroll
  for (int v = 0; v < 4; ++v) f.u[v]     = *(const u32*)(p + 2 * v);
#pragma unroll
  for (int v = 0; v < 4; ++v) f.u[4 + v] = *(const u32*)(p + 16 + 2 * v);
  return f.v;
}

// fragment from LDS-resident f32 tile (attention probabilities)
__device__ __forceinline__ bf16x16 load_frag_lds(const float* base, int ld,
                                                 int k0, int lane) {
  int m = lane & 15, half = lane >> 4;
  const float* p = base + (size_t)m * ld + k0 + 8 * half;
  FragU f;
#pragma unroll
  for (int v = 0; v < 4; ++v) f.u[v]     = pack_bf2(p[2 * v], p[2 * v + 1]);
#pragma unroll
  for (int v = 0; v < 4; ++v) f.u[4 + v] = pack_bf2(p[16 + 2 * v], p[17 + 2 * v]);
  return f.v;
}

__device__ __forceinline__ floatx8 wmma_bf16(bf16x16 a, bf16x16 b, floatx8 c) {
  return __builtin_amdgcn_wmma_f32_16x16x32_bf16(false, a, false, b,
                                                 (short)0, c, false, false);
}

// ---------- CDNA5 async global->LDS copy (16B, tracked on ASYNCcnt) ----------
__device__ __forceinline__ void cp_async16(void* lds_dst, const void* gsrc) {
  u32 l = (u32)(uintptr_t)lds_dst;                 // low 32 bits = LDS offset
  u64 g = (u64)(uintptr_t)gsrc;
  asm volatile("global_load_async_to_lds_b128 %0, %1, off"
               :: "v"(l), "v"(g) : "memory");
}
__device__ __forceinline__ void wait_async0() {
  asm volatile("s_wait_asynccnt 0x0" ::: "memory");
}

// ---------- kernel 0: init amax ----------
__global__ void k_init(u32* amax) { *amax = 0u; }

// ---------- kernel 1: abs-max reduction ----------
__global__ __launch_bounds__(256) void k_amax(const float* __restrict__ x,
                                              u32* amax, int n) {
  float m = 0.f;
  for (int i = blockIdx.x * blockDim.x + threadIdx.x; i < n;
       i += gridDim.x * blockDim.x)
    m = fmaxf(m, fabsf(x[i]));
#pragma unroll
  for (int off = 16; off; off >>= 1) m = fmaxf(m, __shfl_xor(m, off, 32));
  if ((threadIdx.x & 31) == 0) atomicMax(amax, __float_as_uint(m)); // x>=0: uint order == float order
}

// ---------- kernel 2: fake-quant x -> bf16 ----------
__global__ __launch_bounds__(256) void k_quant(const float* __restrict__ x,
                                               const u32* amax,
                                               u16* __restrict__ xq, int n) {
  float scale = fmaxf(__uint_as_float(*amax) / 127.f, 1e-8f);
  for (int i = blockIdx.x * blockDim.x + threadIdx.x; i < n;
       i += gridDim.x * blockDim.x) {
    float q = fminf(fmaxf(rintf(x[i] / scale), -128.f), 127.f) * scale;
    xq[i] = f2bf(q);
  }
}

// ---------- kernel 3: transpose-convert weight [E][P] -> bf16 [P][E] ----------
__global__ __launch_bounds__(256) void k_wt(const float* __restrict__ W,
                                            u16* __restrict__ Wt) {
  int i = blockIdx.x * blockDim.x + threadIdx.x;       // i = e*P + p
  if (i < EE * PP) {
    int e = i / PP, p = i - e * PP;
    Wt[(size_t)p * EE + e] = f2bf(W[i]);
  }
}

// ---------- kernel 4: bf16 GEMM with async-LDS double buffering ----------
// workgroup tile 128x64, 8 waves (4x2), each wave 32x32 = 2x2 WMMA tiles.
// K staged in 64-wide blocks via GLOBAL_LOAD_ASYNC_TO_LDS_B128 (ping-pong).
// mode 0: Q layout [B*H,N,D]  mode 1: K layout  mode 2: V^T [B*H,D,N]
// mode 3: plain f32 [M,768] (final output)
#define LDA 72   // 64 k-elements + 8 pad (144B rows, 16B aligned, bank-spread)
__global__ __launch_bounds__(256) void k_gemm(const u16* __restrict__ A,
                                              const u16* __restrict__ Bt,
                                              const float* __restrict__ bias,
                                              void* __restrict__ outp,
                                              int Kdim, int mode) {
  __shared__ u16 sA[2][128][LDA];   // 36 KB
  __shared__ u16 sB[2][64][LDA];    // 18 KB

  int t = threadIdx.x;
  int lane = t & 31, w = t >> 5;
  int warp_r = w >> 1, warp_c = w & 1;
  int row0 = blockIdx.x * 128;
  int col0 = blockIdx.y * 64;
  int half = lane >> 4, n = lane & 15;

  // stage K-block [kb, kb+64) into buffer `buf`
  auto prefetch = [&](int buf, int kb) {
    // A tile: 128 rows x 128B = 1024 x 16B chunks -> 4 per thread
#pragma unroll
    for (int i = 0; i < 4; ++i) {
      int f = t + i * 256;                 // 16B chunk id
      int r = f >> 3, c = f & 7;
      cp_async16(&sA[buf][r][c * 8],
                 A + (size_t)(row0 + r) * Kdim + kb + c * 8);
    }
    // B tile: 64 rows x 128B = 512 x 16B chunks -> 2 per thread
#pragma unroll
    for (int i = 0; i < 2; ++i) {
      int f = t + i * 256;
      int r = f >> 3, c = f & 7;
      cp_async16(&sB[buf][r][c * 8],
                 Bt + (size_t)(col0 + r) * Kdim + kb + c * 8);
    }
  };

  prefetch(0, 0);
  wait_async0();
  __syncthreads();

  floatx8 acc[2][2] = {};
  int buf = 0;
  for (int kb = 0; kb < Kdim; kb += 64) {
    if (kb + 64 < Kdim) prefetch(buf ^ 1, kb + 64);   // overlap with WMMAs
#pragma unroll
    for (int ks = 0; ks < 64; ks += 32) {
      bf16x16 a0 = load_frag_l(&sA[buf][warp_r * 32][0],      LDA, ks, lane);
      bf16x16 a1 = load_frag_l(&sA[buf][warp_r * 32 + 16][0], LDA, ks, lane);
      bf16x16 b0 = load_frag_l(&sB[buf][warp_c * 32][0],      LDA, ks, lane);
      bf16x16 b1 = load_frag_l(&sB[buf][warp_c * 32 + 16][0], LDA, ks, lane);
      acc[0][0] = wmma_bf16(a0, b0, acc[0][0]);
      acc[0][1] = wmma_bf16(a0, b1, acc[0][1]);
      acc[1][0] = wmma_bf16(a1, b0, acc[1][0]);
      acc[1][1] = wmma_bf16(a1, b1, acc[1][1]);
    }
    wait_async0();        // our copies into buf^1 have landed
    __syncthreads();      // everyone done reading buf / writing buf^1
    buf ^= 1;
  }

#pragma unroll
  for (int tr = 0; tr < 2; ++tr)
#pragma unroll
    for (int tc = 0; tc < 2; ++tc)
#pragma unroll
      for (int r = 0; r < 8; ++r) {
        int grow = row0 + warp_r * 32 + tr * 16 + r + 8 * half;
        int gcol = col0 + warp_c * 32 + tc * 16 + n;
        float v = acc[tr][tc][r] + bias[gcol];
        if (mode == 3) {
          ((float*)outp)[(size_t)grow * EE + gcol] = v;
        } else {
          int b = grow >> 10, nn = grow & 1023;
          int h = gcol >> 6,  d = gcol & 63;
          size_t idx = (mode == 2)
              ? ((size_t)(b * HH + h) * DD + d) * NN + nn
              : ((size_t)(b * HH + h) * NN + nn) * DD + d;
          ((u16*)outp)[idx] = f2bf(v);
        }
      }
}

// ---------- kernel 5: fused attention for one (head, 16-row tile) ----------
// 8 waves; wave w owns key columns [w*128, w*128+128).
__global__ __launch_bounds__(256) void k_attn(const u16* __restrict__ Q,
                                              const u16* __restrict__ K,
                                              const u16* __restrict__ Vt,
                                              u16* __restrict__ ctx) {
  __shared__ float s_log[16 * NN];   // 64 KB logits tile (reused as ctx accum)
  __shared__ float s_rowsum[16];

  int bh = blockIdx.x >> 6;                   // 0..95
  int row0 = (blockIdx.x & 63) << 4;          // query row tile
  const u16* Qb = Q + ((size_t)bh * NN + row0) * DD;
  const u16* Kb = K + (size_t)bh * NN * DD;
  const u16* Vb = Vt + (size_t)bh * DD * NN;

  int lane = threadIdx.x & 31, w = threadIdx.x >> 5;
  int half = lane >> 4, n = lane & 15;

  // ---- phase 1: logits = Q K^T ----
  bf16x16 a0 = load_frag_g(Qb, DD, 0, 0, lane);
  bf16x16 a1 = load_frag_g(Qb, DD, 0, 32, lane);
  for (int jt = w * 8; jt < w * 8 + 8; ++jt) {
    bf16x16 b0 = load_frag_g(Kb, DD, jt * 16, 0, lane);
    bf16x16 b1 = load_frag_g(Kb, DD, jt * 16, 32, lane);
    floatx8 c = {};
    c = wmma_bf16(a0, b0, c);
    c = wmma_bf16(a1, b1, c);
#pragma unroll
    for (int r = 0; r < 8; ++r)
      s_log[(r + 8 * half) * NN + jt * 16 + n] = c[r];
  }
  __syncthreads();

  // ---- phase 2: ITA integer-approx softmax (unnormalized p in LDS) ----
  int row = threadIdx.x >> 4;                 // 0..15, 16 threads per row
  float* rowp = s_log + row * NN + (threadIdx.x & 15) * 64;
  float mx = -1e30f;
  for (int i = 0; i < 64; ++i) mx = fmaxf(mx, rowp[i]);
#pragma unroll
  for (int off = 8; off; off >>= 1) mx = fmaxf(mx, __shfl_xor(mx, off, 16));
  float zmax = mx * LOG2E_F;
  float sum = 0.f;
  for (int i = 0; i < 64; ++i) {
    float z = rowp[i] * LOG2E_F - zmax;       // z <= 0
    float zi = floorf(z);
    float p = exp2f(zi) * (1.f + (z - zi));   // 2^f ~= 1+f
    rowp[i] = p;
    sum += p;
  }
#pragma unroll
  for (int off = 8; off; off >>= 1) sum += __shfl_xor(sum, off, 16);
  if ((threadIdx.x & 15) == 0) s_rowsum[row] = sum;
  __syncthreads();

  // ---- phase 3: ctx partial = P V over this wave's 128 keys ----
  floatx8 cacc[4] = {};
  for (int ch = 0; ch < 4; ++ch) {
    bf16x16 ap = load_frag_lds(s_log, NN, w * 128 + ch * 32, lane);
#pragma unroll
    for (int tc = 0; tc < 4; ++tc) {
      bf16x16 bv = load_frag_g(Vb, NN, tc * 16, w * 128 + ch * 32, lane);
      cacc[tc] = wmma_bf16(ap, bv, cacc[tc]);
    }
  }
  __syncthreads();                            // done reading s_log

  // ---- cross-wave reduction into reused LDS (16x64 f32) ----
  float* s_ctx = s_log;
  for (int i = threadIdx.x; i < 16 * DD; i += 256) s_ctx[i] = 0.f;
  __syncthreads();
#pragma unroll
  for (int tc = 0; tc < 4; ++tc)
#pragma unroll
    for (int r = 0; r < 8; ++r)
      atomicAdd(&s_ctx[(r + 8 * half) * DD + tc * 16 + n], cacc[tc][r]);
  __syncthreads();

  // ---- phase 4: normalize (fold 1/rowsum) and emit bf16 ctx [8192,768] ----
  int b = bh / HH, h = bh - b * HH;
  for (int i = threadIdx.x; i < 16 * DD; i += 256) {
    int rl = i >> 6, d = i & 63;
    float v = s_ctx[i] / s_rowsum[rl];
    ctx[(size_t)(b * NN + row0 + rl) * PP + h * DD + d] = f2bf(v);
  }
}

// ---------- host ----------
extern "C" void kernel_launch(void* const* d_in, const int* in_sizes, int n_in,
                              void* d_out, int out_size, void* d_ws, size_t ws_size,
                              hipStream_t stream) {
  const float* x  = (const float*)d_in[0];
  const float* Wq = (const float*)d_in[1];
  const float* bq = (const float*)d_in[2];
  const float* Wk = (const float*)d_in[3];
  const float* bk = (const float*)d_in[4];
  const float* Wv = (const float*)d_in[5];
  const float* bv = (const float*)d_in[6];
  const float* Wo = (const float*)d_in[7];
  const float* bo = (const float*)d_in[8];

  char* ws   = (char*)d_ws;
  u32*  amax = (u32*)(ws + OFF_AMAX);
  u16*  XQ   = (u16*)(ws + OFF_XQ);
  u16*  WQT  = (u16*)(ws + OFF_WQT);
  u16*  WKT  = (u16*)(ws + OFF_WKT);
  u16*  WVT  = (u16*)(ws + OFF_WVT);
  u16*  WOT  = (u16*)(ws + OFF_WOT);
  u16*  Qb   = (u16*)(ws + OFF_Q);
  u16*  Kb   = (u16*)(ws + OFF_K);
  u16*  VTb  = (u16*)(ws + OFF_VT);
  u16*  CTX  = (u16*)(ws + OFF_CTX);

  const int nX = BN * EE;
  const int nW = EE * PP;

  k_init<<<1, 1, 0, stream>>>(amax);
  k_amax<<<1024, 256, 0, stream>>>(x, amax, nX);
  k_quant<<<(nX + 255) / 256, 256, 0, stream>>>(x, amax, XQ, nX);
  k_wt<<<(nW + 255) / 256, 256, 0, stream>>>(Wq, WQT);
  k_wt<<<(nW + 255) / 256, 256, 0, stream>>>(Wk, WKT);
  k_wt<<<(nW + 255) / 256, 256, 0, stream>>>(Wv, WVT);
  k_wt<<<(nW + 255) / 256, 256, 0, stream>>>(Wo, WOT);

  dim3 gg(BN / 128, PP / 64);
  k_gemm<<<gg, 256, 0, stream>>>(XQ, WQT, bq, (void*)Qb,  EE, 0);
  k_gemm<<<gg, 256, 0, stream>>>(XQ, WKT, bk, (void*)Kb,  EE, 1);
  k_gemm<<<gg, 256, 0, stream>>>(XQ, WVT, bv, (void*)VTb, EE, 2);

  k_attn<<<BB * HH * (NN / 16), 256, 0, stream>>>(Qb, Kb, VTb, CTX);

  k_gemm<<<gg, 256, 0, stream>>>(CTX, WOT, bo, d_out, PP, 3);
}
